// ProposalModule_65120294142371
// MI455X (gfx1250) — matrix operand order repaired and checked by reading, not compiled, not run on hardware
//
#include <hip/hip_runtime.h>
#include <hip/hip_bf16.h>
#include <stdint.h>

typedef __attribute__((ext_vector_type(16))) __bf16 v16bf;
typedef __attribute__((ext_vector_type(8)))  __bf16 v8bf;
typedef __attribute__((ext_vector_type(8)))  float  v8f;

#define B_      32
#define K_      2048
#define C_      256
#define NPROP   256
#define NSAMPLE 16
#define NSC     18
#define NCLS    18
#define OUTCH   119   // 2+3+24+72+18
#define OUTPAD  128
#define KP1     288   // 259 padded to 9*32
#define LDC     288   // LDS column stride (elements)
#define PB      4     // proposals per SA workgroup (64 columns -> 4 N-tiles)
#define RADIUS2 0.09f

__device__ __forceinline__ unsigned short f2bf(float f) {
    union { float f; unsigned int u; } v; v.f = f;
    unsigned int u = v.u;
    u += 0x7FFFu + ((u >> 16) & 1u);   // round-to-nearest-even
    return (unsigned short)(u >> 16);
}
__device__ __forceinline__ float bf2f(unsigned short h) {
    union { unsigned int u; float f; } v; v.u = ((unsigned int)h) << 16;
    return v.f;
}

// ---------------------------------------------------------------- prep
__global__ void prep_kernel(
    const float* __restrict__ sa_w1, const float* __restrict__ sa_w2, const float* __restrict__ sa_w3,
    const float* __restrict__ sa_gamma, const float* __restrict__ sa_beta,
    const float* __restrict__ sa_mean,  const float* __restrict__ sa_var,
    const float* __restrict__ p_w1, const float* __restrict__ p_w2, const float* __restrict__ p_w3,
    const float* __restrict__ p_b3,
    const float* __restrict__ p_gamma, const float* __restrict__ p_beta,
    const float* __restrict__ p_mean,  const float* __restrict__ p_var,
    unsigned short* __restrict__ w1bf, unsigned short* __restrict__ w2bf, unsigned short* __restrict__ w3bf,
    unsigned short* __restrict__ pw1bf, unsigned short* __restrict__ pw2bf, unsigned short* __restrict__ pw3bf,
    float* __restrict__ sa_scale, float* __restrict__ sa_bias,
    float* __restrict__ p_scale,  float* __restrict__ p_bias, float* __restrict__ pb3pad)
{
    int i = blockIdx.x * 256 + threadIdx.x;
    if (i < 256 * KP1) { int r = i / KP1, k = i % KP1;
        w1bf[i] = f2bf(k < 259 ? sa_w1[r * 259 + k] : 0.f); return; }
    i -= 256 * KP1;
    if (i < 65536) { w2bf[i]  = f2bf(sa_w2[i]); return; } i -= 65536;
    if (i < 65536) { w3bf[i]  = f2bf(sa_w3[i]); return; } i -= 65536;
    if (i < 65536) { pw1bf[i] = f2bf(p_w1[i]);  return; } i -= 65536;
    if (i < 65536) { pw2bf[i] = f2bf(p_w2[i]);  return; } i -= 65536;
    if (i < OUTPAD * 256) { int r = i / 256, k = i % 256;
        pw3bf[i] = f2bf(r < OUTCH ? p_w3[r * 256 + k] : 0.f); return; }
    i -= OUTPAD * 256;
    if (i < 3 * 256) { float sc = sa_gamma[i] * rsqrtf(sa_var[i] + 1e-5f);
        sa_scale[i] = sc; sa_bias[i] = sa_beta[i] - sa_mean[i] * sc; return; }
    i -= 3 * 256;
    if (i < 2 * 256) { float sc = p_gamma[i] * rsqrtf(p_var[i] + 1e-5f);
        p_scale[i] = sc; p_bias[i] = p_beta[i] - p_mean[i] * sc; return; }
    i -= 2 * 256;
    if (i < OUTPAD) { pb3pad[i] = (i < OUTCH) ? p_b3[i] : 0.f; return; }
}

// ---------------------------------------------------------------- FPS
__global__ __launch_bounds__(256) void fps_kernel(
    const float* __restrict__ xyz, int* __restrict__ inds, float* __restrict__ newxyz)
{
    int b = blockIdx.x, tid = threadIdx.x;
    __shared__ float dist[K_];
    __shared__ float rv[256];
    __shared__ int   ri[256];
    __shared__ int   s_far;
    __shared__ float s_c[3];
    const float* px = xyz + (size_t)b * K_ * 3;
    for (int k = tid; k < K_; k += 256) dist[k] = 1e10f;
    if (tid == 0) s_far = 0;
    __syncthreads();
    for (int it = 0; it < NPROP; ++it) {
        int far = s_far;
        if (tid == 0) {
            inds[b * NPROP + it] = far;
            float cx = px[far*3], cy = px[far*3+1], cz = px[far*3+2];
            s_c[0] = cx; s_c[1] = cy; s_c[2] = cz;
            float* nx = newxyz + ((size_t)b * NPROP + it) * 3;
            nx[0] = cx; nx[1] = cy; nx[2] = cz;
        }
        __syncthreads();
        float cx = s_c[0], cy = s_c[1], cz = s_c[2];
        float bv = -1.f; int bi = 0x7fffffff;
        for (int k = tid; k < K_; k += 256) {
            float dx = px[k*3]-cx, dy = px[k*3+1]-cy, dz = px[k*3+2]-cz;
            float d  = fminf(dist[k], dx*dx + dy*dy + dz*dz);
            dist[k] = d;
            if (d > bv || (d == bv && k < bi)) { bv = d; bi = k; }
        }
        rv[tid] = bv; ri[tid] = bi;
        __syncthreads();
        for (int s = 128; s > 0; s >>= 1) {
            if (tid < s) {
                float ov = rv[tid + s]; int oi = ri[tid + s];
                if (ov > rv[tid] || (ov == rv[tid] && oi < ri[tid])) { rv[tid] = ov; ri[tid] = oi; }
            }
            __syncthreads();
        }
        if (tid == 0) s_far = ri[0];
        __syncthreads();
    }
}

// ---------------------------------------------------------------- ball query (one wave32 per proposal)
__global__ __launch_bounds__(256) void ballq_kernel(
    const float* __restrict__ xyz, const float* __restrict__ newxyz, int* __restrict__ bidx)
{
    int wave = threadIdx.x >> 5, lane = threadIdx.x & 31;
    int b = blockIdx.y, p = blockIdx.x * 8 + wave;
    const float* px = xyz + (size_t)b * K_ * 3;
    const float* c  = newxyz + ((size_t)b * NPROP + p) * 3;
    float cx = c[0], cy = c[1], cz = c[2];
    int found = 0;
    int my[NSAMPLE];
    for (int base = 0; base < K_ && found < NSAMPLE; base += 32) {
        int k = base + lane;
        float dx = px[k*3]-cx, dy = px[k*3+1]-cy, dz = px[k*3+2]-cz;
        bool within = (dx*dx + dy*dy + dz*dz) < RADIUS2;
        unsigned mask = (unsigned)__ballot(within);   // wave32: low 32 bits
        while (mask && found < NSAMPLE) {
            int pos = __builtin_ctz(mask);
            mask &= mask - 1;
            my[found++] = base + pos;                 // uniform across lanes
        }
    }
    if (lane == 0) {
        int first = (found > 0) ? my[0] : 0;
        int* out = bidx + ((size_t)b * NPROP + p) * NSAMPLE;
        for (int j = 0; j < NSAMPLE; ++j) out[j] = (j < found) ? my[j] : first;
    }
}

// ---------------------------------------------------------------- WMMA layer (A-fragment reused across NT N-tiles)
// A (16x32 bf16): lane<16 -> row=lane, K kt*32+0..15 ; lane>=16 -> row=lane-16, K kt*32+16..31
// B (32x16 bf16): lane<16 -> col=lane, K kt*32+0..15 ; lane>=16 -> col=lane-16, K kt*32+16..31
// C/D (16x16 f32): lane<16 -> vgpr i = M i ; lane>=16 -> vgpr i = M 8+i ; N = lane&15
template<int MT, int NT>
__device__ __forceinline__ void mlp_layer(
    const unsigned short* __restrict__ Xin, unsigned short* __restrict__ Xout,
    const unsigned short* __restrict__ W, int KP, int nkt,
    const float* __restrict__ scale, const float* __restrict__ bias, int mt0, int lane)
{
    const int nloc = lane & 15;
    const int khB  = (lane >> 4) << 4;
    const int koA  = (lane < 16) ? 0 : 16;
    const int r    = lane & 15;
    v8f zero = {};
    v8f acc[MT][NT];
    #pragma unroll
    for (int m = 0; m < MT; ++m)
        #pragma unroll
        for (int n = 0; n < NT; ++n) acc[m][n] = zero;
    for (int kt = 0; kt < nkt; ++kt) {
        v16bf af[MT];
        #pragma unroll
        for (int m = 0; m < MT; ++m) {
            const unsigned short* wp = W + (size_t)((mt0 + m) * 16 + r) * KP + kt * 32 + koA;
            __builtin_prefetch(wp, 0, 0);
            union { v16bf v; v8bf h[2]; } ua;
            ua.h[0] = *(const v8bf*)(wp);
            ua.h[1] = *(const v8bf*)(wp + 8);
            af[m] = ua.v;
        }
        #pragma unroll
        for (int n = 0; n < NT; ++n) {
            v16bf bf = *(const v16bf*)(Xin + (n * 16 + nloc) * LDC + kt * 32 + khB);
            #pragma unroll
            for (int m = 0; m < MT; ++m)
                acc[m][n] = __builtin_amdgcn_wmma_f32_16x16x32_bf16(
                    false, af[m], false, bf, (short)0, acc[m][n], false, false);
        }
    }
    #pragma unroll
    for (int m = 0; m < MT; ++m) {
        const int chb = (mt0 + m) * 16 + ((lane < 16) ? 0 : 8);
        #pragma unroll
        for (int n = 0; n < NT; ++n) {
            const int col = n * 16 + nloc;
            #pragma unroll
            for (int i = 0; i < 8; ++i) {
                int ch = chb + i;
                float v = fmaxf(acc[m][n][i] * scale[ch] + bias[ch], 0.f);  // fused BN + ReLU
                Xout[col * LDC + ch] = f2bf(v);
            }
        }
    }
}

template<int NT>
__device__ __forceinline__ void final_layer(
    const unsigned short* __restrict__ Xin, const unsigned short* __restrict__ W,
    const float* __restrict__ pb3, float* __restrict__ netbase, int mt, int lane)
{
    const int nloc = lane & 15;
    const int khB  = (lane >> 4) << 4;
    const int koA  = (lane < 16) ? 0 : 16;
    const int r    = lane & 15;
    v8f zero = {};
    v8f acc[NT];
    #pragma unroll
    for (int n = 0; n < NT; ++n) acc[n] = zero;
    for (int kt = 0; kt < 8; ++kt) {
        const unsigned short* wp = W + (size_t)(mt * 16 + r) * 256 + kt * 32 + koA;
        union { v16bf v; v8bf h[2]; } ua;
        ua.h[0] = *(const v8bf*)(wp);
        ua.h[1] = *(const v8bf*)(wp + 8);
        #pragma unroll
        for (int n = 0; n < NT; ++n) {
            v16bf bf = *(const v16bf*)(Xin + (n * 16 + nloc) * LDC + kt * 32 + khB);
            acc[n] = __builtin_amdgcn_wmma_f32_16x16x32_bf16(
                false, ua.v, false, bf, (short)0, acc[n], false, false);
        }
    }
    const int chb = mt * 16 + ((lane < 16) ? 0 : 8);
    #pragma unroll
    for (int n = 0; n < NT; ++n) {
        const int col = n * 16 + nloc;
        #pragma unroll
        for (int i = 0; i < 8; ++i)
            netbase[(size_t)col * OUTPAD + chb + i] = acc[n][i] + pb3[chb + i];
    }
}

// ---------------------------------------------------------------- SA MLP: one WG per (b, 4 proposals)
__global__ __launch_bounds__(256) void sa_kernel(
    const float* __restrict__ xyz, const float* __restrict__ feat,
    const float* __restrict__ newxyz, const int* __restrict__ bidx,
    const unsigned short* __restrict__ w1, const unsigned short* __restrict__ w2,
    const unsigned short* __restrict__ w3,
    const float* __restrict__ sa_scale, const float* __restrict__ sa_bias,
    float* __restrict__ y_ws)
{
    const int NCOL = PB * NSAMPLE;                       // 64 columns
    int p0 = blockIdx.x * PB, b = blockIdx.y, tid = threadIdx.x;
    __shared__ __align__(32) unsigned short bufA[NCOL * LDC];
    __shared__ __align__(32) unsigned short bufB[NCOL * LDC];
    __shared__ int   s_idx[NCOL];
    __shared__ float s_c[PB * 3];
    if (tid < NCOL)   s_idx[tid] = bidx[((size_t)b * NPROP + p0) * NSAMPLE + tid];
    if (tid < PB * 3) s_c[tid]   = newxyz[((size_t)b * NPROP + p0) * 3 + tid];
    __syncthreads();
    const float* px = xyz  + (size_t)b * K_ * 3;
    const float* pf = feat + (size_t)b * C_ * K_;
    for (int e = tid; e < NCOL * LDC; e += 256) {        // X: column-major bf16, 288 rows
        int col = e / LDC, row = e % LDC;
        int pp = col >> 4;
        int g  = s_idx[col];
        float v;
        if      (row < 3)   v = (px[g * 3 + row] - s_c[pp * 3 + row]) * (1.0f / 0.3f);
        else if (row < 259) v = pf[(size_t)(row - 3) * K_ + g];
        else                v = 0.f;
        bufA[col * LDC + row] = f2bf(v);
    }
    __syncthreads();
    int wave = tid >> 5, lane = tid & 31;
    mlp_layer<2, PB>(bufA, bufB, w1, KP1, 9, sa_scale,       sa_bias,       wave * 2, lane);
    __syncthreads();
    mlp_layer<2, PB>(bufB, bufA, w2, 256, 8, sa_scale + 256, sa_bias + 256, wave * 2, lane);
    __syncthreads();
    mlp_layer<2, PB>(bufA, bufB, w3, 256, 8, sa_scale + 512, sa_bias + 512, wave * 2, lane);
    __syncthreads();
    #pragma unroll
    for (int pp = 0; pp < PB; ++pp) {                    // max over samples
        float m = -1e30f;
        for (int j = 0; j < NSAMPLE; ++j)
            m = fmaxf(m, bf2f(bufB[(pp * NSAMPLE + j) * LDC + tid]));
        y_ws[((size_t)b * C_ + tid) * NPROP + p0 + pp] = m;
    }
}

// ---------------------------------------------------------------- head: one WG per (b, 32 props)
__global__ __launch_bounds__(256) void head_kernel(
    const float* __restrict__ y_ws,
    const unsigned short* __restrict__ pw1, const unsigned short* __restrict__ pw2,
    const unsigned short* __restrict__ pw3,
    const float* __restrict__ p_scale, const float* __restrict__ p_bias,
    const float* __restrict__ pb3, float* __restrict__ net_ws)
{
    int pg = blockIdx.x, b = blockIdx.y, tid = threadIdx.x;
    __shared__ __align__(32) unsigned short bufA[32 * LDC];
    __shared__ __align__(32) unsigned short bufB[32 * LDC];
    int p0 = pg * 32;
    const float* yp = y_ws + ((size_t)b * C_ + tid) * NPROP + p0;
    for (int n = 0; n < 32; ++n) bufA[n * LDC + tid] = f2bf(yp[n]);
    __syncthreads();
    int wave = tid >> 5, lane = tid & 31;
    mlp_layer<2, 2>(bufA, bufB, pw1, 256, 8, p_scale,       p_bias,       wave * 2, lane);
    __syncthreads();
    mlp_layer<2, 2>(bufB, bufA, pw2, 256, 8, p_scale + 256, p_bias + 256, wave * 2, lane);
    __syncthreads();
    float* netbase = net_ws + ((size_t)b * NPROP + p0) * OUTPAD;
    final_layer<2>(bufA, pw3, pb3, netbase, wave, lane);
}

// ---------------------------------------------------------------- decode
__global__ __launch_bounds__(256) void decode_kernel(
    const float* __restrict__ net, const float* __restrict__ newxyz,
    const float* __restrict__ msa, float* __restrict__ out)
{
    const int NP = B_ * NPROP;
    int gid = blockIdx.x * 256 + threadIdx.x;
    if (gid >= NP) return;
    const float* nt = net + (size_t)gid * OUTPAD;
    const float* c0 = newxyz + (size_t)gid * 3;

    float* o_obj = out;
    float* o_cen = o_obj + (size_t)NP * 2;
    float* o_ss  = o_cen + (size_t)NP * 3;
    float* o_sr  = o_ss  + (size_t)NP * NSC;
    float* o_ps  = o_sr  + (size_t)NP * NSC * 3;
    float* o_sem = o_ps  + (size_t)NP * 3;
    float* o_cor = o_sem + (size_t)NP * NSC;
    float* o_sl  = o_cor + (size_t)NP * 24;
    float* o_op  = o_sl  + (size_t)NP * (NCLS + 1);
    float* o_sp  = o_op  + (size_t)NP;

    float obj0 = nt[0], obj1 = nt[1];
    o_obj[gid * 2 + 0] = obj0; o_obj[gid * 2 + 1] = obj1;
    float cx = c0[0] + nt[2], cy = c0[1] + nt[3], cz = c0[2] + nt[4];
    o_cen[gid * 3 + 0] = cx; o_cen[gid * 3 + 1] = cy; o_cen[gid * 3 + 2] = cz;

    int best = 0; float bestv = -1e30f;
    for (int j = 0; j < NSC; ++j) {
        float ss = nt[29 + j];
        o_ss[gid * NSC + j] = ss;
        if (ss > bestv) { bestv = ss; best = j; }
    }
    float ps[3];
    for (int j = 0; j < NSC; ++j)
        for (int r = 0; r < 3; ++r) {
            float sr = nt[47 + j * 3 + r] * msa[j * 3 + r];
            o_sr[gid * NSC * 3 + j * 3 + r] = sr;
            if (j == best) ps[r] = sr + msa[j * 3 + r];
        }
    for (int r = 0; r < 3; ++r) o_ps[gid * 3 + r] = ps[r];

    float smax = -1e30f;
    for (int j = 0; j < NCLS; ++j) { float s = nt[101 + j]; o_sem[gid * NCLS + j] = s; smax = fmaxf(smax, s); }

    const float sx[8] = {1,1,-1,-1,1,1,-1,-1};
    const float sy[8] = {1,1,1,1,-1,-1,-1,-1};
    const float sz[8] = {1,-1,-1,1,1,-1,-1,1};
    float ccx = cx, ccy = cz, ccz = -cy;     // angle == 0 -> R = I
    for (int k = 0; k < 8; ++k) {
        o_cor[gid * 24 + k * 3 + 0] = ccx + sx[k] * ps[0] * 0.5f;
        o_cor[gid * 24 + k * 3 + 1] = ccy + sy[k] * ps[2] * 0.5f;
        o_cor[gid * 24 + k * 3 + 2] = ccz + sz[k] * ps[1] * 0.5f;
    }
    for (int j = 0; j < NCLS; ++j) o_sl[gid * (NCLS + 1) + j] = nt[101 + j];
    o_sl[gid * (NCLS + 1) + NCLS] = (obj0 <= obj1 ? 0.f : 1.f) * 1e10f;

    float m = fmaxf(obj0, obj1);
    float e0 = __expf(obj0 - m), e1 = __expf(obj1 - m);
    o_op[gid] = e1 / (e0 + e1);

    float ssum = 0.f;
    for (int j = 0; j < NCLS; ++j) ssum += __expf(nt[101 + j] - smax);
    float inv = 1.f / ssum;
    for (int j = 0; j < NCLS; ++j) o_sp[gid * NCLS + j] = __expf(nt[101 + j] - smax) * inv;
}

// ---------------------------------------------------------------- host
extern "C" void kernel_launch(void* const* d_in, const int* in_sizes, int n_in,
                              void* d_out, int out_size, void* d_ws, size_t ws_size,
                              hipStream_t stream) {
    const float* xyz      = (const float*)d_in[0];
    const float* features = (const float*)d_in[1];
    const float* sa_w1    = (const float*)d_in[2];
    const float* sa_w2    = (const float*)d_in[3];
    const float* sa_w3    = (const float*)d_in[4];
    const float* sa_gamma = (const float*)d_in[5];
    const float* sa_beta  = (const float*)d_in[6];
    const float* sa_mean  = (const float*)d_in[7];
    const float* sa_var   = (const float*)d_in[8];
    const float* p_w1     = (const float*)d_in[9];
    const float* p_w2     = (const float*)d_in[10];
    const float* p_w3     = (const float*)d_in[11];
    const float* p_b3     = (const float*)d_in[12];
    const float* p_gamma  = (const float*)d_in[13];
    const float* p_beta   = (const float*)d_in[14];
    const float* p_mean   = (const float*)d_in[15];
    const float* p_var    = (const float*)d_in[16];
    const float* msa      = (const float*)d_in[17];

    uint8_t* ws = (uint8_t*)d_ws;
    size_t off = 0;
    auto take = [&](size_t nbytes) -> void* {
        void* p = ws + off;
        off = (off + nbytes + 63) & ~(size_t)63;
        return p;
    };
    int*   inds    = (int*)  take((size_t)B_ * NPROP * 4);
    float* newxyz  = (float*)take((size_t)B_ * NPROP * 3 * 4);
    int*   ballidx = (int*)  take((size_t)B_ * NPROP * NSAMPLE * 4);
    float* y_ws    = (float*)take((size_t)B_ * C_ * NPROP * 4);
    float* net_ws  = (float*)take((size_t)B_ * NPROP * OUTPAD * 4);
    unsigned short* w1bf  = (unsigned short*)take((size_t)256 * KP1 * 2);
    unsigned short* w2bf  = (unsigned short*)take((size_t)256 * 256 * 2);
    unsigned short* w3bf  = (unsigned short*)take((size_t)256 * 256 * 2);
    unsigned short* pw1bf = (unsigned short*)take((size_t)256 * 256 * 2);
    unsigned short* pw2bf = (unsigned short*)take((size_t)256 * 256 * 2);
    unsigned short* pw3bf = (unsigned short*)take((size_t)OUTPAD * 256 * 2);
    float* sa_scale = (float*)take(3 * 256 * 4);
    float* sa_bias  = (float*)take(3 * 256 * 4);
    float* p_scale  = (float*)take(2 * 256 * 4);
    float* p_bias   = (float*)take(2 * 256 * 4);
    float* pb3pad   = (float*)take(OUTPAD * 4);

    int prep_elems = 256 * KP1 + 4 * 65536 + OUTPAD * 256 + 3 * 256 + 2 * 256 + OUTPAD;
    prep_kernel<<<(prep_elems + 255) / 256, 256, 0, stream>>>(
        sa_w1, sa_w2, sa_w3, sa_gamma, sa_beta, sa_mean, sa_var,
        p_w1, p_w2, p_w3, p_b3, p_gamma, p_beta, p_mean, p_var,
        w1bf, w2bf, w3bf, pw1bf, pw2bf, pw3bf,
        sa_scale, sa_bias, p_scale, p_bias, pb3pad);

    fps_kernel<<<B_, 256, 0, stream>>>(xyz, inds, newxyz);

    ballq_kernel<<<dim3(NPROP / 8, B_), 256, 0, stream>>>(xyz, newxyz, ballidx);

    sa_kernel<<<dim3(NPROP / PB, B_), 256, 0, stream>>>(
        xyz, features, newxyz, ballidx, w1bf, w2bf, w3bf, sa_scale, sa_bias, y_ws);

    head_kernel<<<dim3(NPROP / 32, B_), 256, 0, stream>>>(
        y_ws, pw1bf, pw2bf, pw3bf, p_scale, p_bias, pb3pad, net_ws);

    decode_kernel<<<(B_ * NPROP + 255) / 256, 256, 0, stream>>>(net_ws, newxyz, msa, (float*)d_out);
}
